// RecurrentAttention_40973988004104
// MI455X (gfx1250) — matrix operand order repaired and checked by reading, not compiled, not run on hardware
//
#include <hip/hip_runtime.h>
#include <math.h>

// ---- CDNA5 (gfx1250) types ---------------------------------------------------
typedef __attribute__((ext_vector_type(16))) __bf16    v16bf;
typedef __attribute__((ext_vector_type(8)))  float     v8f;
typedef __attribute__((ext_vector_type(4)))  unsigned  v4u;
typedef __attribute__((ext_vector_type(8)))  int       v8i_;
typedef __attribute__((ext_vector_type(4)))  int       v4i_;

#define BATCH   4
#define CDIM    192
#define NHEADS  4
#define HW      16384      // 128*128
#define TCH     1152       // 576 (qkv) + 192 (q_mut) + 384 (kv_mut)

#define LDSOFF(p) ((unsigned)(uintptr_t)(const void*)(p))

// =============================================================================
// Tensor Data Mover: async 2-D tile (bf16) global -> LDS.
// D# group0: [1:0]=count=1, [63:32]=lds_addr, [120:64]=global byte addr,
//            [127:126]=type 2.  group1: data_size=1 (2B), tensor dims set huge
//            (no OOB clipping), tile_dim0 = contiguous elts, tile_dim1 = rows,
//            tensor_dim0_stride = row stride (elts).  Tracked by TENSORcnt.
// =============================================================================
__device__ __forceinline__ void tdm_load_2d(unsigned lds_byte, const void* gaddr,
                                            unsigned tile_d0, unsigned tile_d1,
                                            unsigned long long stride0)
{
    const unsigned long long ga = (unsigned long long)(uintptr_t)gaddr;
    const unsigned td0 = 1u << 30, td1 = 1u << 20;    // generous tensor dims
    v4u g0;
    g0.x = 1u;                                               // count=1, user load
    g0.y = lds_byte;                                         // lds_addr
    g0.z = (unsigned)ga;                                     // global_addr[31:0]
    g0.w = (unsigned)((ga >> 32) & 0x01ffffffu) | (2u << 30);// [56:32] | type=2
    v8i_ g1;
    g1[0] = (int)(1u << 16);                                 // data_size=1 (2B)
    g1[1] = (int)((td0 & 0xffffu) << 16);                    // tensor_dim0[15:0]
    g1[2] = (int)(((td0 >> 16) & 0xffffu) | ((td1 & 0xffffu) << 16));
    g1[3] = (int)(((td1 >> 16) & 0xffffu) | ((tile_d0 & 0xffffu) << 16));
    g1[4] = (int)(tile_d1 & 0xffffu);                        // tile_dim1 (tile_dim2=0)
    g1[5] = (int)(unsigned)(stride0 & 0xffffffffull);        // stride0[31:0]
    g1[6] = (int)(unsigned)((stride0 >> 32) & 0xffffull);    // stride0[47:32]
    g1[7] = 0;
    v4i_ z4 = {0, 0, 0, 0};
#if defined(__clang_major__) && (__clang_major__ >= 23)
    v8i_ z8 = {0, 0, 0, 0, 0, 0, 0, 0};
    __builtin_amdgcn_tensor_load_to_lds(g0, g1, z4, z4, z8, 0);
#else
    __builtin_amdgcn_tensor_load_to_lds(g0, g1, z4, z4, 0);
#endif
}

// =============================================================================
// f32 -> bf16 convert (weights + activations)
// =============================================================================
__global__ __launch_bounds__(256)
void cvt_f32_bf16(const float* __restrict__ in, __bf16* __restrict__ out, long long n)
{
    long long i = (long long)blockIdx.x * 256 + threadIdx.x;
    if (i < n) out[i] = (__bf16)in[i];
}

// =============================================================================
// bf16 WMMA GEMM: C[M,N] = A[M,K] * B[K,N]   (row-major, f32 accum)
// block = 256 threads (8 waves), tile M=64, N=128, K-step 32.
// TDM double-buffered LDS staging: wave 0 issues tile s+1 while all waves
// compute tile s;  s_wait_tensorcnt(2) retires tile s (TDM is in-order).
// =============================================================================
template<bool OUT_F32>
__global__ __launch_bounds__(256)
void gemm_bf16_wmma(const __bf16* __restrict__ A, int lda, long long sA,
                    const __bf16* __restrict__ Bm, long long sB,
                    void* __restrict__ Cvoid, long long sC,
                    int K, int N, int accumulate)
{
    const int z = blockIdx.z;
    A  += z * sA;
    Bm += z * sB;
    const int m0 = blockIdx.y * 64;
    const int n0 = blockIdx.x * 128;

    __shared__ __align__(16) __bf16 As[2][64][32];
    __shared__ __align__(16) __bf16 Bs[2][32][128];

    const int tid  = threadIdx.x;
    const int lane = tid & 31;
    const int wv   = tid >> 5;
    const int mw   = wv & 3;
    const int nw   = wv >> 2;

    v8f acc[4] = {};

    const int nst = K / 32;
    auto issue = [&](int s, int buf) {
        tdm_load_2d(LDSOFF(&As[buf][0][0]), A + (long long)m0 * lda + s * 32,
                    32, 64, (unsigned long long)lda);
        tdm_load_2d(LDSOFF(&Bs[buf][0][0]), Bm + (long long)(s * 32) * N + n0,
                    128, 32, (unsigned long long)N);
    };

    if (wv == 0) issue(0, 0);

    for (int s = 0; s < nst; s++) {
        if (wv == 0) {
            if (s + 1 < nst) { issue(s + 1, (s + 1) & 1); __builtin_amdgcn_s_wait_tensorcnt(2); }
            else             { __builtin_amdgcn_s_wait_tensorcnt(0); }
        }
        __syncthreads();

        const int buf = s & 1;
        v16bf a = *(const v16bf*)&As[buf][mw * 16 + (lane & 15)][(lane >> 4) * 16];
        #pragma unroll
        for (int nt = 0; nt < 4; nt++) {
            v16bf b;
            #pragma unroll
            for (int s2 = 0; s2 < 16; s2++) b[s2] = Bs[buf][(lane >> 4) * 16 + s2][nw * 64 + nt * 16 + (lane & 15)];
            acc[nt] = __builtin_amdgcn_wmma_f32_16x16x32_bf16(
                false, a, false, b, (short)0, acc[nt], false, false);
        }
        __syncthreads();
    }

    // C/D VGPR layout: VGPR r: lanes 0-15 -> M=r, lanes 16-31 -> M=8+r; N=lane%16
    if (OUT_F32) {
        float* C = (float*)Cvoid + z * sC;
        #pragma unroll
        for (int nt = 0; nt < 4; nt++) {
            #pragma unroll
            for (int r = 0; r < 8; r++) {
                int row = m0 + mw * 16 + (lane >> 4) * 8 + r;
                int col = n0 + nw * 64 + nt * 16 + (lane & 15);
                long long o = (long long)row * N + col;
                float v = acc[nt][r];
                if (accumulate) v += C[o];
                C[o] = v;
            }
        }
    } else {
        __bf16* C = (__bf16*)Cvoid + z * sC;
        #pragma unroll
        for (int nt = 0; nt < 4; nt++) {
            #pragma unroll
            for (int r = 0; r < 8; r++) {
                int row = m0 + mw * 16 + (lane >> 4) * 8 + r;
                int col = n0 + nw * 64 + nt * 16 + (lane & 15);
                C[(long long)row * N + col] = (__bf16)acc[nt][r];
            }
        }
    }
}

// =============================================================================
// depthwise 3x3, SAME padding, bf16 in/out, f32 math. Channel -> weight map:
// [0,576): qkv_dw  [576,768): q_mut_dw  [768,1152): kv_mut_dw
// =============================================================================
__global__ __launch_bounds__(256)
void dwconv3(const __bf16* __restrict__ src, __bf16* __restrict__ dst,
             const float* __restrict__ wqkv, const float* __restrict__ wqm,
             const float* __restrict__ wkvm)
{
    long long idx = (long long)blockIdx.x * 256 + threadIdx.x;
    const long long total = (long long)BATCH * TCH * HW;
    if (idx >= total) return;
    int n  = (int)(idx & (HW - 1));
    int ch = (int)((idx >> 14) % TCH);
    int b  = (int)(idx / ((long long)TCH * HW));
    int y = n >> 7, x = n & 127;

    const float* wp;
    if (ch < 576)      wp = wqkv + ch * 9;
    else if (ch < 768) wp = wqm  + (ch - 576) * 9;
    else               wp = wkvm + (ch - 768) * 9;

    const __bf16* s = src + ((long long)b * TCH + ch) * HW;
    float acc = 0.f;
    #pragma unroll
    for (int ky = 0; ky < 3; ky++) {
        int yy = y + ky - 1;
        if ((unsigned)yy >= 128u) continue;
        #pragma unroll
        for (int kx = 0; kx < 3; kx++) {
            int xx = x + kx - 1;
            if ((unsigned)xx >= 128u) continue;
            acc += wp[ky * 3 + kx] * (float)s[yy * 128 + xx];
        }
    }
    dst[idx] = (__bf16)acc;
}

// =============================================================================
// Gram kernel: per (grp, b, h) compute G = M * M^T, M = [Q;K] is 96 x 16384.
// blockIdx.x = grp*16 + b*4 + h.
// TDM double-buffered staging of 96x128 bf16 tiles (two 48-row channel runs,
// row stride HW).  288 threads = 9 waves in a 3x3 grid; each wave owns a
// static 2x2 tile block (32x32 of 96x96) -> 4 named v8f accumulators.
// Diagonal of G carries the l2 norms -> l2norm fused away from big tensors.
// =============================================================================
__global__ __launch_bounds__(288)
void gram_wmma(const __bf16* __restrict__ t2, float* __restrict__ gram)
{
    const int grp = blockIdx.x >> 4;
    const int bh  = blockIdx.x & 15;
    const int b   = bh >> 2, h = bh & 3;

    __shared__ __align__(16) __bf16 Ms[2][96][128];

    const int tid  = threadIdx.x;
    const int lane = tid & 31;
    const int wv   = tid >> 5;            // 0..8
    const int twr  = wv / 3;              // 0..2  (row block of 32)
    const int twc  = wv % 3;              // 0..2  (col block of 32)

    const int r0   = twr * 32 + (lane & 15);
    const int c0   = twc * 32 + (lane & 15);
    const int ksub = (lane >> 4) * 16;

    const int chq = (grp ? 576 : 0)   + h * 48;   // q / q_mut block base channel
    const int chk = (grp ? 720 : 144) + h * 48;   // k / k_mut block base channel
    const __bf16* gq = t2 + (long long)(b * TCH + chq) * HW;
    const __bf16* gk = t2 + (long long)(b * TCH + chk) * HW;

    v8f acc00 = {}, acc01 = {}, acc10 = {}, acc11 = {};

    const int NSTEP = HW / 128;           // 128 K-steps of 128
    auto issue = [&](int s, int buf) {
        tdm_load_2d(LDSOFF(&Ms[buf][0][0]),  gq + s * 128, 128, 48, (unsigned long long)HW);
        tdm_load_2d(LDSOFF(&Ms[buf][48][0]), gk + s * 128, 128, 48, (unsigned long long)HW);
    };

    if (wv == 0) issue(0, 0);

    for (int s = 0; s < NSTEP; s++) {
        if (wv == 0) {
            if (s + 1 < NSTEP) { issue(s + 1, (s + 1) & 1); __builtin_amdgcn_s_wait_tensorcnt(2); }
            else               { __builtin_amdgcn_s_wait_tensorcnt(0); }
        }
        __syncthreads();

        const __bf16 (*M)[128] = Ms[s & 1];
        #pragma unroll
        for (int kk = 0; kk < 128; kk += 32) {
            v16bf a0 = *(const v16bf*)&M[r0     ][kk + ksub];
            v16bf a1 = *(const v16bf*)&M[r0 + 16][kk + ksub];
            v16bf b0 = *(const v16bf*)&M[c0     ][kk + ksub];
            v16bf b1 = *(const v16bf*)&M[c0 + 16][kk + ksub];
            acc00 = __builtin_amdgcn_wmma_f32_16x16x32_bf16(false, a0, false, b0, (short)0, acc00, false, false);
            acc01 = __builtin_amdgcn_wmma_f32_16x16x32_bf16(false, a0, false, b1, (short)0, acc01, false, false);
            acc10 = __builtin_amdgcn_wmma_f32_16x16x32_bf16(false, a1, false, b0, (short)0, acc10, false, false);
            acc11 = __builtin_amdgcn_wmma_f32_16x16x32_bf16(false, a1, false, b1, (short)0, acc11, false, false);
        }
        __syncthreads();
    }

    // store 2x2 tiles; C layout: VGPR r: lanes 0-15 -> M=r, lanes 16-31 -> M=8+r
    float* G = gram + (long long)blockIdx.x * 96 * 96;
    const int orow = twr * 32 + (lane >> 4) * 8;
    const int ocol = twc * 32 + (lane & 15);
    #pragma unroll
    for (int r = 0; r < 8; r++) {
        G[(orow + r) * 96      + ocol     ] = acc00[r];
        G[(orow + r) * 96      + ocol + 16] = acc01[r];
        G[(orow + r + 16) * 96 + ocol     ] = acc10[r];
        G[(orow + r + 16) * 96 + ocol + 16] = acc11[r];
    }
}

// =============================================================================
// Softmax of normalized Gram cross-block + fold proj through it:
// P1 = proj_L[:, h*48:+48] @ attn_h ; P2 = proj_R[:, h*48:+48] @ cross_h
// Pcat[b][192][384] bf16:  cols [h*48, +48) <- P1, cols [192+h*48, +48) <- P2
// blockIdx.x = b*4 + h
// =============================================================================
__global__ __launch_bounds__(256)
void softmax_compose(const float* __restrict__ gram,
                     const float* __restrict__ proj_w,
                     const float* __restrict__ temp,
                     const float* __restrict__ temp_mut,
                     __bf16* __restrict__ Pcat)
{
    const int bid = blockIdx.x;
    const int b = bid >> 2, h = bid & 3;
    __shared__ float smat[2][48][48];
    const int tid = threadIdx.x;

    if (tid < 96) {
        int grp = tid / 48, c = tid % 48;
        const float* G = gram + (long long)(grp * 16 + bid) * 9216;
        float T  = grp ? temp_mut[h] : temp[h];
        float nq = fmaxf(sqrtf(fmaxf(G[c * 96 + c], 0.f)), 1e-12f);
        float row[48];
        float mx = -3.4e38f;
        for (int d = 0; d < 48; d++) {
            int dd = 48 + d;
            float nk = fmaxf(sqrtf(fmaxf(G[dd * 96 + dd], 0.f)), 1e-12f);
            float sv = G[c * 96 + dd] / (nq * nk) * T;
            row[d] = sv;
            mx = fmaxf(mx, sv);
        }
        float ssum = 0.f;
        for (int d = 0; d < 48; d++) { float e = __expf(row[d] - mx); row[d] = e; ssum += e; }
        float inv = 1.f / ssum;
        for (int d = 0; d < 48; d++) smat[grp][c][d] = row[d] * inv;
    }
    __syncthreads();

    for (int e = tid; e < 2 * 192 * 48; e += 256) {
        int grp = e / 9216, rem = e % 9216;
        int o = rem / 48, d = rem % 48;
        const float* pr = proj_w + o * 384 + grp * 192 + h * 48;
        float s = 0.f;
        for (int c = 0; c < 48; c++) s += pr[c] * smat[grp][c][d];
        Pcat[((long long)b * 192 + o) * 384 + grp * 192 + h * 48 + d] = (__bf16)s;
    }
}

// =============================================================================
// host launch
// =============================================================================
extern "C" void kernel_launch(void* const* d_in, const int* in_sizes, int n_in,
                              void* d_out, int out_size, void* d_ws, size_t ws_size,
                              hipStream_t stream)
{
    const float* x        = (const float*)d_in[0];
    const float* x_hidden = (const float*)d_in[1];
    const float* qkv_w    = (const float*)d_in[2];
    const float* qkv_dw   = (const float*)d_in[3];
    const float* qm_w     = (const float*)d_in[4];
    const float* qm_dw    = (const float*)d_in[5];
    const float* kvm_w    = (const float*)d_in[6];
    const float* kvm_dw   = (const float*)d_in[7];
    const float* proj_w   = (const float*)d_in[8];
    const float* temp     = (const float*)d_in[9];
    const float* temp_mut = (const float*)d_in[10];

    // workspace layout (all offsets 256B aligned)
    char* p = (char*)d_ws;
    __bf16* Wx   = (__bf16*)p; p += 768 * 192 * 2;                 // qkv_w ++ q_mut_w
    __bf16* Wh   = (__bf16*)p; p += 384 * 192 * 2;                 // kv_mut_w
    __bf16* xb   = (__bf16*)p; p += (long long)BATCH * CDIM * HW * 2;
    __bf16* hb   = (__bf16*)p; p += (long long)BATCH * CDIM * HW * 2;
    __bf16* t1   = (__bf16*)p; p += (long long)BATCH * TCH * HW * 2;  // conv1x1 out
    __bf16* t2   = (__bf16*)p; p += (long long)BATCH * TCH * HW * 2;  // dwconv out
    float*  gram = (float*)p;  p += 2LL * BATCH * NHEADS * 96 * 96 * 4;
    __bf16* Pcat = (__bf16*)p; p += (long long)BATCH * 192 * 384 * 2;

    auto nb = [](long long n) { return (unsigned)((n + 255) / 256); };

    // 1) bf16 conversions
    cvt_f32_bf16<<<nb(576 * 192), 256, 0, stream>>>(qkv_w, Wx, 576 * 192);
    cvt_f32_bf16<<<nb(192 * 192), 256, 0, stream>>>(qm_w, Wx + 576 * 192, 192 * 192);
    cvt_f32_bf16<<<nb(384 * 192), 256, 0, stream>>>(kvm_w, Wh, 384 * 192);
    cvt_f32_bf16<<<nb((long long)BATCH * CDIM * HW), 256, 0, stream>>>(x, xb, (long long)BATCH * CDIM * HW);
    cvt_f32_bf16<<<nb((long long)BATCH * CDIM * HW), 256, 0, stream>>>(x_hidden, hb, (long long)BATCH * CDIM * HW);

    // 2) conv1x1 GEMMs (WMMA bf16):  t1[b, 0:768)  = Wx @ x_b ;  t1[b, 768:1152) = Wh @ xh_b
    {
        dim3 g(HW / 128, 768 / 64, BATCH);
        gemm_bf16_wmma<false><<<g, 256, 0, stream>>>(
            Wx, 192, 0LL, xb, (long long)CDIM * HW,
            (void*)t1, (long long)TCH * HW, 192, HW, 0);
    }
    {
        dim3 g(HW / 128, 384 / 64, BATCH);
        gemm_bf16_wmma<false><<<g, 256, 0, stream>>>(
            Wh, 192, 0LL, hb, (long long)CDIM * HW,
            (void*)(t1 + (long long)768 * HW), (long long)TCH * HW, 192, HW, 0);
    }

    // 3) depthwise 3x3
    dwconv3<<<nb((long long)BATCH * TCH * HW), 256, 0, stream>>>(t1, t2, qkv_dw, qm_dw, kvm_dw);

    // 4) Gram matrices (fused l2norm), 32 groups = {self,mut} x B x HEADS
    gram_wmma<<<32, 288, 0, stream>>>(t2, gram);

    // 5) softmax + fold proj through attn -> Pcat[b][192][384]
    softmax_compose<<<BATCH * NHEADS, 256, 0, stream>>>(gram, proj_w, temp, temp_mut, Pcat);

    // 6) final GEMM (WMMA bf16, f32 out): out_b = P1_b @ v_b + P2_b @ vmut_b
    {
        dim3 g(HW / 128, 192 / 64, BATCH);
        gemm_bf16_wmma<true><<<g, 256, 0, stream>>>(
            Pcat, 384, (long long)192 * 384, t2 + (long long)384 * HW, (long long)TCH * HW,
            d_out, (long long)CDIM * HW, 192, HW, 0);
        gemm_bf16_wmma<true><<<g, 256, 0, stream>>>(
            Pcat + 192, 384, (long long)192 * 384, t2 + (long long)960 * HW, (long long)TCH * HW,
            d_out, (long long)CDIM * HW, 192, HW, 1);
    }
}
